// DeltaResidualExpanded_28724741276185
// MI455X (gfx1250) — compile-verified, matrix operand order checked by reading, not run on hardware
//
#include <hip/hip_runtime.h>

// Problem constants from the reference
#define B_    4
#define T_    2048
#define D_    1024
#define DV_   8
#define NTOK  (B_*T_)
#define TPB   256      // 8 wave32s per block
#define TOKPB 16       // tokens per block (M dim of the WMMA tile)
#define EPS_K_    1e-6f
#define EPS_NORM_ 1e-6f

typedef __attribute__((ext_vector_type(16))) _Float16 v16h;
typedef __attribute__((ext_vector_type(8)))  float    v8f;

__launch_bounds__(TPB)
__global__ void delta_residual_kernel(const float* __restrict__ X,
                                      const float* __restrict__ sub,   // sublayer_output
                                      const float* __restrict__ xin,   // x_in
                                      const float* __restrict__ gnw,   // gate_norm_w (D)
                                      const float* __restrict__ gw,    // gate_w (1,D)
                                      const float* __restrict__ gb,    // gate_b (1)
                                      const float* __restrict__ Wv,    // (DV,D)
                                      float* __restrict__ out) {
  __shared__ float s_kscale[TOKPB];
  __shared__ float s_beta[TOKPB];
  __shared__ float s_v[TOKPB * 16];   // WMMA C tile: [token m][col n], n<8 valid
  __shared__ float s_kTX[DV_];

  const int tid  = threadIdx.x;
  const int lane = tid & 31;
  const int wave = tid >> 5;
  const int tokBase = blockIdx.x * TOKPB;

  // ---------------- Phase 1a: per-token scalars (2 tokens per wave) --------
  #pragma unroll
  for (int s = 0; s < 2; ++s) {
    const int t16 = wave * 2 + s;
    const int tok = tokBase + t16;
    const float* subp = sub + (size_t)tok * D_;
    const float* xp   = xin + (size_t)tok * D_;
    float ssk = 0.f, ssx = 0.f, dg = 0.f;
    for (int d = lane; d < D_; d += 32) {
      float kk = subp[d]; ssk = fmaf(kk, kk, ssk);
      float xx = xp[d];   ssx = fmaf(xx, xx, ssx);
      dg = fmaf(xx, gnw[d] * gw[d], dg);
    }
    #pragma unroll
    for (int off = 16; off > 0; off >>= 1) {
      ssk += __shfl_xor(ssk, off, 32);
      ssx += __shfl_xor(ssx, off, 32);
      dg  += __shfl_xor(dg,  off, 32);
    }
    if (lane == 0) {
      float nrm = sqrtf(ssk);
      s_kscale[t16] = 1.0f / (fmaxf(nrm, EPS_K_) * sqrtf((float)D_));
      float rms   = rsqrtf(ssx * (1.0f / (float)D_) + EPS_NORM_);
      float logit = rms * dg + gb[0];
      s_beta[t16] = 2.0f / (1.0f + __expf(-logit));
    }
  }

  s_v[tid] = 0.0f;   // 256 entries cleared by 256 threads
  __syncthreads();

  // ------- Phase 1b: v = x_in @ Wv^T via v_wmma_f32_16x16x32_f16 ----------
  // M = 16 tokens of this block, N = 16 (cols 0..7 = Wv rows, 8..15 zero pad),
  // K = D split across 8 waves (128 each => 4 WMMAs of K=32 per wave).
  // Per-lane fragment = two contiguous 8-float runs -> four b128 loads each.
  {
    const int mn   = lane & 15;          // M (A) / N (B) index for this lane
    const int hi   = lane >> 4;          // lane group
    const int koff = hi * 8;
    const float* xrow  = xin + (size_t)(tokBase + mn) * D_;
    const float* wrowp = Wv + (size_t)(mn & 7) * D_;   // clamped: always valid
    const float  wmask = (mn < DV_) ? 1.0f : 0.0f;     // branch-free zero pad
    v8f c = {};
    #pragma unroll
    for (int kb = 0; kb < 4; ++kb) {
      const int d0 = wave * 128 + kb * 32 + koff;
      // A: x_in[token mn, d0 .. d0+7] and [d0+16 .. d0+23]
      float4 a0 = *(const float4*)(xrow + d0);
      float4 a1 = *(const float4*)(xrow + d0 + 4);
      float4 a2 = *(const float4*)(xrow + d0 + 16);
      float4 a3 = *(const float4*)(xrow + d0 + 20);
      // B: Wv[row mn, same K runs], masked for pad columns
      float4 b0 = *(const float4*)(wrowp + d0);
      float4 b1 = *(const float4*)(wrowp + d0 + 4);
      float4 b2 = *(const float4*)(wrowp + d0 + 16);
      float4 b3 = *(const float4*)(wrowp + d0 + 20);

      v16h a, b;
      a[0]=(_Float16)a0.x; a[1]=(_Float16)a0.y; a[2]=(_Float16)a0.z; a[3]=(_Float16)a0.w;
      a[4]=(_Float16)a1.x; a[5]=(_Float16)a1.y; a[6]=(_Float16)a1.z; a[7]=(_Float16)a1.w;
      a[8]=(_Float16)a2.x; a[9]=(_Float16)a2.y; a[10]=(_Float16)a2.z; a[11]=(_Float16)a2.w;
      a[12]=(_Float16)a3.x; a[13]=(_Float16)a3.y; a[14]=(_Float16)a3.z; a[15]=(_Float16)a3.w;
      b[0]=(_Float16)(wmask*b0.x); b[1]=(_Float16)(wmask*b0.y);
      b[2]=(_Float16)(wmask*b0.z); b[3]=(_Float16)(wmask*b0.w);
      b[4]=(_Float16)(wmask*b1.x); b[5]=(_Float16)(wmask*b1.y);
      b[6]=(_Float16)(wmask*b1.z); b[7]=(_Float16)(wmask*b1.w);
      b[8]=(_Float16)(wmask*b2.x); b[9]=(_Float16)(wmask*b2.y);
      b[10]=(_Float16)(wmask*b2.z); b[11]=(_Float16)(wmask*b2.w);
      b[12]=(_Float16)(wmask*b3.x); b[13]=(_Float16)(wmask*b3.y);
      b[14]=(_Float16)(wmask*b3.z); b[15]=(_Float16)(wmask*b3.w);

      c = __builtin_amdgcn_wmma_f32_16x16x32_f16(
              /*neg_a=*/false, a, /*neg_b=*/false, b,
              /*c_mod=*/(short)0, c, /*reuse_a=*/false, /*reuse_b=*/false);
    }
    // combine K-partials across waves: ds_add_f32 into the shared C tile
    #pragma unroll
    for (int r = 0; r < 8; ++r) {
      const int m = r + hi * 8;          // C/D layout: VGPR r -> M = r (+8)
      atomicAdd(&s_v[m * 16 + mn], c[r]);
    }
  }
  __syncthreads();

  // ---------------- Phase 2: single-pass stream of X per token ------------
  for (int t16 = 0; t16 < TOKPB; ++t16) {
    const int tok = tokBase + t16;
    if (tid < DV_) s_kTX[tid] = 0.0f;
    __syncthreads();

    const float kscale = s_kscale[t16];
    const float beta   = s_beta[t16];
    const float* subp  = sub + (size_t)tok * D_;
    const float4* Xt   = (const float4*)(X   + (size_t)tok * (D_ * DV_));
    float4*       Ot   = (float4*)      (out + (size_t)tok * (D_ * DV_));

    // Pull the next token's 32KB X tile toward the caches (global_prefetch_b8):
    // 256 threads x 128B stride covers the whole tile.
    if (t16 + 1 < TOKPB) {
      const char* nxt = (const char*)(X + (size_t)(tok + 1) * (D_ * DV_));
      __builtin_prefetch(nxt + tid * 128, 0, 0);
    }

    float4 xlo[4], xhi[4];
    float  kd[4];
    float  part[DV_] = {0.f, 0.f, 0.f, 0.f, 0.f, 0.f, 0.f, 0.f};
    #pragma unroll
    for (int i = 0; i < 4; ++i) {
      const int d = i * TPB + tid;               // coalesced 32B per lane
      kd[i]  = subp[d] * kscale;                 // k[d] (already scaled)
      xlo[i] = Xt[2 * d];
      xhi[i] = Xt[2 * d + 1];
      part[0] = fmaf(kd[i], xlo[i].x, part[0]);
      part[1] = fmaf(kd[i], xlo[i].y, part[1]);
      part[2] = fmaf(kd[i], xlo[i].z, part[2]);
      part[3] = fmaf(kd[i], xlo[i].w, part[3]);
      part[4] = fmaf(kd[i], xhi[i].x, part[4]);
      part[5] = fmaf(kd[i], xhi[i].y, part[5]);
      part[6] = fmaf(kd[i], xhi[i].z, part[6]);
      part[7] = fmaf(kd[i], xhi[i].w, part[7]);
    }
    // wave32 tree reduce, then one lane per wave combines in LDS
    #pragma unroll
    for (int j = 0; j < DV_; ++j) {
      #pragma unroll
      for (int off = 16; off > 0; off >>= 1)
        part[j] += __shfl_xor(part[j], off, 32);
    }
    if (lane == 0) {
      #pragma unroll
      for (int j = 0; j < DV_; ++j) atomicAdd(&s_kTX[j], part[j]);
    }
    __syncthreads();

    float corr[DV_];
    #pragma unroll
    for (int j = 0; j < DV_; ++j) corr[j] = s_v[t16 * 16 + j] - s_kTX[j];

    #pragma unroll
    for (int i = 0; i < 4; ++i) {
      const int d = i * TPB + tid;
      const float bk = beta * kd[i];
      float4 olo, ohi;
      olo.x = fmaf(bk, corr[0], xlo[i].x);
      olo.y = fmaf(bk, corr[1], xlo[i].y);
      olo.z = fmaf(bk, corr[2], xlo[i].z);
      olo.w = fmaf(bk, corr[3], xlo[i].w);
      ohi.x = fmaf(bk, corr[4], xhi[i].x);
      ohi.y = fmaf(bk, corr[5], xhi[i].y);
      ohi.z = fmaf(bk, corr[6], xhi[i].z);
      ohi.w = fmaf(bk, corr[7], xhi[i].w);
      Ot[2 * d]     = olo;
      Ot[2 * d + 1] = ohi;
    }
    __syncthreads();   // protect s_kTX before next token zeroes it
  }
}

extern "C" void kernel_launch(void* const* d_in, const int* in_sizes, int n_in,
                              void* d_out, int out_size, void* d_ws, size_t ws_size,
                              hipStream_t stream) {
  const float* X   = (const float*)d_in[0];
  const float* sub = (const float*)d_in[1];
  const float* xin = (const float*)d_in[2];
  const float* gnw = (const float*)d_in[3];
  const float* gw  = (const float*)d_in[4];
  const float* gb  = (const float*)d_in[5];
  const float* Wv  = (const float*)d_in[6];
  float* out = (float*)d_out;

  dim3 grid(NTOK / TOKPB);   // 512 blocks
  dim3 block(TPB);           // 256 threads = 8 wave32
  delta_residual_kernel<<<grid, block, 0, stream>>>(X, sub, xin, gnw, gw, gb, Wv, out);
}
